// WaveNetworkLM_77283641524868
// MI455X (gfx1250) — matrix operand order repaired and checked by reading, not compiled, not run on hardware
//
#include <hip/hip_runtime.h>
#include <cstddef>
#include <cstdint>

// ---------------------------------------------------------------------------
// Types for CDNA5 WMMA (wave32, 16x16x32 bf16 -> f32)
// ---------------------------------------------------------------------------
typedef __attribute__((ext_vector_type(16))) __bf16 bf16x16;
typedef __attribute__((ext_vector_type(8)))  __bf16 bf16x8;
typedef __attribute__((ext_vector_type(4)))  __bf16 bf16x4;
typedef __attribute__((ext_vector_type(8)))  float  floatx8;

#define EPSF 1e-6f

#define EPI_NONE    0
#define EPI_GELU    1
#define EPI_SIGMOID 2

// GEMM tiling
#define BM 128
#define BN 128
#define BK 32
#define LDT 40                 // LDS row stride in bf16 (32 data + 8 pad = 80B)
#define LDT_BYTES 80
#define TILE_BYTES (BM * LDT_BYTES)   // 10240B per tile buffer

// ---------------------------------------------------------------------------
// Tiled GEMM, bf16 operands, B pre-transposed in global memory (Bt = [N,K]).
//   C = epi( A[M,K]@B[K,N] (+ A2[M,K2]@B2[K2,N]) + biases )
// Both A and B tiles are staged with global_load_async_to_lds_b128 into
// double-buffered LDS (padded 80B rows); completion via s_wait_asynccnt.
// 256 threads = 8 waves; wave grid 4(M) x 2(N); wave tile 32x64 = 2x4 WMMA.
// ---------------------------------------------------------------------------
__global__ __launch_bounds__(256) void gemm_bf16_kernel(
    const __bf16* __restrict__ A,   const __bf16* __restrict__ Bt,
    const __bf16* __restrict__ A2,  const __bf16* __restrict__ Bt2,
    const float* __restrict__ bias1, const float* __restrict__ bias2,
    const float* __restrict__ bias3,
    void* __restrict__ Cout, int M, int N, int K, int K2, int epi, int outBf)
{
    __shared__ __bf16 As[2][BM][LDT];   // [buf][m][k]
    __shared__ __bf16 Bs[2][BN][LDT];   // [buf][n][k]

    const int tid   = threadIdx.x;
    const int lane  = tid & 31;
    const int wave  = tid >> 5;
    const int waveM = (wave >> 1) * 32;   // 4 waves along M
    const int waveN = (wave & 1) * 64;    // 2 waves along N

    const unsigned asBase = (unsigned)(uintptr_t)&As[0][0][0];
    const unsigned bsBase = (unsigned)(uintptr_t)&Bs[0][0][0];

    floatx8 acc[2][4];
    #pragma unroll
    for (int mi = 0; mi < 2; ++mi)
        #pragma unroll
        for (int ni = 0; ni < 4; ++ni)
            acc[mi][ni] = (floatx8){0.f,0.f,0.f,0.f,0.f,0.f,0.f,0.f};

    union Frag { bf16x16 v; bf16x8 h[2]; };

    // issue one tile's async copies: 2x16B for A + 2x16B for B per thread
    auto issue_tile = [&](int buf, const __bf16* Ag, const __bf16* Bg, int Kp) {
        unsigned aB = asBase + (unsigned)buf * TILE_BYTES;
        unsigned bB = bsBase + (unsigned)buf * TILE_BYTES;
        #pragma unroll
        for (int it = 0; it < 2; ++it) {
            int idx = tid + 256 * it;        // 0..511 chunks of 16B
            int row = idx >> 2;              // 4x16B per 64B data row
            int seg = idx & 3;
            const __bf16* ga = Ag + (size_t)row * Kp + seg * 8;
            unsigned la = aB + (unsigned)(row * LDT_BYTES + seg * 16);
            asm volatile("global_load_async_to_lds_b128 %0, %1, off"
                         :: "v"(la), "v"(ga) : "memory");
        }
        #pragma unroll
        for (int it = 0; it < 2; ++it) {
            int idx = tid + 256 * it;
            int row = idx >> 2;
            int seg = idx & 3;
            const __bf16* gb = Bg + (size_t)row * Kp + seg * 8;
            unsigned lb = bB + (unsigned)(row * LDT_BYTES + seg * 16);
            asm volatile("global_load_async_to_lds_b128 %0, %1, off"
                         :: "v"(lb), "v"(gb) : "memory");
        }
    };

    for (int pass = 0; pass < 2; ++pass) {
        const __bf16* Ap  = pass ? A2  : A;
        const __bf16* Btp = pass ? Bt2 : Bt;
        const int     Kp  = pass ? K2  : K;
        if (Ap == nullptr || Kp == 0) break;

        const __bf16* Agbase = Ap  + (size_t)blockIdx.y * BM * Kp;
        const __bf16* Bgbase = Btp + (size_t)blockIdx.x * BN * Kp;
        const int ktiles = Kp / BK;

        issue_tile(0, Agbase, Bgbase, Kp);

        for (int kt = 0; kt < ktiles; ++kt) {
            const int cur = kt & 1;
            const bool more = (kt + 1 < ktiles);
            if (more)
                issue_tile(cur ^ 1, Agbase + (kt + 1) * BK, Bgbase + (kt + 1) * BK, Kp);

            // in-order async completion: 4 outstanding from next tile allowed
            if (more) asm volatile("s_wait_asynccnt 0x4" ::: "memory");
            else      asm volatile("s_wait_asynccnt 0x0" ::: "memory");
            __syncthreads();

            const __bf16 (*Asc)[LDT] = As[cur];
            const __bf16 (*Bsc)[LDT] = Bs[cur];

            // A frag (16x32): lanes 0-15 -> K 0-7 & 16-23; lanes 16-31 -> K 8-15 & 24-31
            Frag af[2];
            {
                int ar = waveM + (lane & 15);
                int ak = (lane >> 4) * 8;
                af[0].h[0] = *reinterpret_cast<const bf16x8*>(&Asc[ar][ak]);
                af[0].h[1] = *reinterpret_cast<const bf16x8*>(&Asc[ar][ak + 16]);
                af[1].h[0] = *reinterpret_cast<const bf16x8*>(&Asc[ar + 16][ak]);
                af[1].h[1] = *reinterpret_cast<const bf16x8*>(&Asc[ar + 16][ak + 16]);
            }
            // B frag (32x16): lanes 0-15 -> K 0-7 & 16-23; lanes 16-31 -> K 8-15 & 24-31
            Frag bfg[4];
            {
                int bc  = waveN + (lane & 15);
                int bk0 = (lane >> 4) * 8;
                #pragma unroll
                for (int ni = 0; ni < 4; ++ni) {
                    bfg[ni].h[0] = *reinterpret_cast<const bf16x8*>(&Bsc[bc + ni * 16][bk0]);
                    bfg[ni].h[1] = *reinterpret_cast<const bf16x8*>(&Bsc[bc + ni * 16][bk0 + 16]);
                }
            }
            // 8 WMMA per wave per K-tile
            #pragma unroll
            for (int mi = 0; mi < 2; ++mi)
                #pragma unroll
                for (int ni = 0; ni < 4; ++ni)
                    acc[mi][ni] = __builtin_amdgcn_wmma_f32_16x16x32_bf16(
                        false, af[mi].v, false, bfg[ni].v,
                        (short)0, acc[mi][ni], false, false);
            __syncthreads();
        }
    }

    // ---- epilogue: bias + activation + store (f32 or bf16) ----
    const int rbase = blockIdx.y * BM + waveM + ((lane >> 4) * 8);
    const int cbase = blockIdx.x * BN + waveN + (lane & 15);
    float*  Cf = (float*)Cout;
    __bf16* Cb = (__bf16*)Cout;
    #pragma unroll
    for (int ni = 0; ni < 4; ++ni) {
        int col = cbase + ni * 16;
        float bsum = 0.f;
        if (bias1) bsum += bias1[col];
        if (bias2) bsum += bias2[col];
        if (bias3) bsum += bias3[col];
        #pragma unroll
        for (int mi = 0; mi < 2; ++mi) {
            #pragma unroll
            for (int i = 0; i < 8; ++i) {
                int row = rbase + mi * 16 + i;
                float z = acc[mi][ni][i] + bsum;
                if (epi == EPI_GELU) {
                    z = 0.5f * z * (1.0f + erff(z * 0.70710678118654752f));
                } else if (epi == EPI_SIGMOID) {
                    z = 1.0f / (1.0f + expf(-z));
                }
                if (outBf) Cb[(size_t)row * N + col] = (__bf16)z;
                else       Cf[(size_t)row * N + col] = z;
            }
        }
    }
}

// ---------------------------------------------------------------------------
// f32 [K,N] -> bf16 [N,K] transposed conversion (weight prep).
// 32x32 tiles through LDS; K,N multiples of 32.
// ---------------------------------------------------------------------------
__global__ __launch_bounds__(256) void k_cvt_bf16_t(
    const float* __restrict__ src, __bf16* __restrict__ dst, int K, int N)
{
    __shared__ __bf16 t[32][33];
    int k0 = blockIdx.y * 32;
    int n0 = blockIdx.x * 32;
    int tx = threadIdx.x & 31;
    int ty = threadIdx.x >> 5;   // 0..7
    #pragma unroll
    for (int j = 0; j < 4; ++j) {
        int k = ty + j * 8;
        t[k][tx] = (__bf16)src[(size_t)(k0 + k) * N + n0 + tx];
    }
    __syncthreads();
    #pragma unroll
    for (int j = 0; j < 4; ++j) {
        int n = ty + j * 8;
        dst[(size_t)(n0 + n) * K + k0 + tx] = t[tx][n];
    }
}

// ---------------------------------------------------------------------------
// Reduction 1: per-(b, s-chunk) partial sums of clip(x)^2 over S
// ---------------------------------------------------------------------------
__global__ __launch_bounds__(256) void k_gsq_partial(
    const float* __restrict__ x, float* __restrict__ partial, int S, int D)
{
    const int SC = 32;
    int b  = blockIdx.x / SC;
    int sc = blockIdx.x % SC;
    int schunk = S / SC;
    const float* xb = x + (size_t)b * S * D;
    float acc[4] = {0.f, 0.f, 0.f, 0.f};
    for (int s = sc * schunk; s < (sc + 1) * schunk; ++s) {
        const float* row = xb + (size_t)s * D;
        #pragma unroll
        for (int j = 0; j < 4; ++j) {
            float v = row[threadIdx.x + j * 256];
            v = fminf(fmaxf(v, -10.f), 10.f);
            acc[j] += v * v;
        }
    }
    #pragma unroll
    for (int j = 0; j < 4; ++j)
        partial[(size_t)blockIdx.x * D + threadIdx.x + j * 256] = acc[j];
}

// ---------------------------------------------------------------------------
// Reduction 2: per-channel G_local[b,d] and global G[b]
// ---------------------------------------------------------------------------
__global__ __launch_bounds__(256) void k_gsq_final(
    const float* __restrict__ partial, float* __restrict__ Gl,
    float* __restrict__ Gg, int D)
{
    const int SC = 32;
    int b = blockIdx.x;
    __shared__ float red[256];
    float tot = 0.f;
    #pragma unroll
    for (int j = 0; j < 4; ++j) {
        int d = threadIdx.x + j * 256;
        float s = 0.f;
        for (int sc = 0; sc < SC; ++sc)
            s += partial[((size_t)b * SC + sc) * D + d];
        Gl[(size_t)b * D + d] = fmaxf(sqrtf(s + EPSF), EPSF);
        tot += s;
    }
    red[threadIdx.x] = tot;
    __syncthreads();
    for (int off = 128; off > 0; off >>= 1) {
        if (threadIdx.x < off) red[threadIdx.x] += red[threadIdx.x + off];
        __syncthreads();
    }
    if (threadIdx.x == 0)
        Gg[b] = fmaxf(sqrtf(red[0] + EPSF), EPSF);
}

// ---------------------------------------------------------------------------
// Wave representation (global + local), combine, amplitude/phase.
// ---------------------------------------------------------------------------
__device__ __forceinline__ void wave2(float xc, float G, float& re, float& im)
{
    float r = floorf(xc / G);                 // torch floor-division semantics
    r = fminf(fmaxf(r, -0.99f), 0.99f);
    float s = sqrtf(fmaxf(1.f - r * r, EPSF));
    float h = sqrtf(r * r + s * s);           // cos/sin of atan2(s, r), s > 0
    re = G * (r / h);
    im = G * (s / h);
}

__global__ __launch_bounds__(256) void k_wave(
    const float* __restrict__ x, const float* __restrict__ Gg,
    const float* __restrict__ Gl,
    float* __restrict__ cr, float* __restrict__ ci,
    __bf16* __restrict__ amp, __bf16* __restrict__ ph, int S, int D)
{
    size_t idx = (size_t)blockIdx.x * blockDim.x + threadIdx.x;
    int d = (int)(idx % D);
    int b = (int)(idx / ((size_t)S * D));
    float xc = fminf(fmaxf(x[idx], -10.f), 10.f);
    float rs, is, rt, it;
    wave2(xc, Gg[b], rs, is);
    wave2(xc, Gl[(size_t)b * D + d], rt, it);
    float crv = rs + rt;
    float civ = is + it;
    cr[idx]  = crv;
    ci[idx]  = civ;
    amp[idx] = (__bf16)sqrtf(crv * crv + civ * civ + EPSF);
    ph[idx]  = (__bf16)atan2f(civ, crv);
}

// gr = cr*gate, gi = ci*gate -> bf16 (GEMM A operands)
__global__ __launch_bounds__(256) void k_gate_mul(
    const float* __restrict__ gate, const float* __restrict__ cr,
    const float* __restrict__ ci,
    __bf16* __restrict__ gr, __bf16* __restrict__ gi)
{
    size_t idx = (size_t)blockIdx.x * blockDim.x + threadIdx.x;
    float g = gate[idx];
    gr[idx] = (__bf16)(cr[idx] * g);
    gi[idx] = (__bf16)(ci[idx] * g);
}

// RMSNorm over last dim (f32 in, bf16 out): one block per row
__global__ __launch_bounds__(256) void k_rmsnorm(
    const float* __restrict__ in, const float* __restrict__ g,
    __bf16* __restrict__ out, int D)
{
    const float* row = in + (size_t)blockIdx.x * D;
    __bf16* orow = out + (size_t)blockIdx.x * D;
    __shared__ float red[256];
    float ss = 0.f;
    #pragma unroll
    for (int j = 0; j < 4; ++j) {
        float v = row[threadIdx.x + j * 256];
        ss += v * v;
    }
    red[threadIdx.x] = ss;
    __syncthreads();
    for (int off = 128; off > 0; off >>= 1) {
        if (threadIdx.x < off) red[threadIdx.x] += red[threadIdx.x + off];
        __syncthreads();
    }
    float scale = rsqrtf(red[0] / (float)D + EPSF);
    #pragma unroll
    for (int j = 0; j < 4; ++j) {
        int d = threadIdx.x + j * 256;
        orow[d] = (__bf16)(row[d] * scale * g[d]);
    }
}

// ---------------------------------------------------------------------------
// Host side
// ---------------------------------------------------------------------------
static inline void gemm(hipStream_t s,
                        const __bf16* A, const __bf16* Bt,
                        const __bf16* A2, const __bf16* Bt2,
                        const float* b1, const float* b2, const float* b3,
                        void* C, int M, int N, int K, int K2, int epi, int outBf)
{
    dim3 grid(N / BN, M / BM), block(256);
    gemm_bf16_kernel<<<grid, block, 0, s>>>(A, Bt, A2, Bt2, b1, b2, b3,
                                            C, M, N, K, K2, epi, outBf);
}

// transpose-convert weight W[K,N] f32 -> Wt[N,K] bf16
static inline void cvtT(hipStream_t s, const float* src, __bf16* dst, int K, int N)
{
    dim3 grid(N / 32, K / 32), block(256);
    k_cvt_bf16_t<<<grid, block, 0, s>>>(src, dst, K, N);
}

extern "C" void kernel_launch(void* const* d_in, const int* in_sizes, int n_in,
                              void* d_out, int out_size, void* d_ws, size_t ws_size,
                              hipStream_t stream)
{
    const int B = 4, S = 4096, D = 1024;
    const int M = B * S;                    // 16384 token rows
    const size_t ND = (size_t)M * D;

    const float* x         = (const float*)d_in[0];
    const float* Wa        = (const float*)d_in[1];
    const float* ba        = (const float*)d_in[2];
    const float* Wp        = (const float*)d_in[3];
    const float* bp        = (const float*)d_in[4];
    const float* gate_bias = (const float*)d_in[5];
    const float* w1r       = (const float*)d_in[6];
    const float* b1r       = (const float*)d_in[7];
    const float* w2r       = (const float*)d_in[8];
    const float* b2r       = (const float*)d_in[9];
    const float* w1i       = (const float*)d_in[10];
    const float* b1i       = (const float*)d_in[11];
    const float* w2i       = (const float*)d_in[12];
    const float* b2i       = (const float*)d_in[13];
    const float* We        = (const float*)d_in[14];
    const float* be        = (const float*)d_in[15];
    const float* g_rms     = (const float*)d_in[16];
    const float* wf1       = (const float*)d_in[17];
    const float* bf1       = (const float*)d_in[18];
    const float* wf2       = (const float*)d_in[19];
    const float* bf2       = (const float*)d_in[20];
    float* y = (float*)d_out;

    // ---- workspace carve-up (256B aligned slices) ----
    char* p = (char*)d_ws;
    auto alloc = [&p](size_t bytes) {
        char* r = p;
        p += (bytes + 255) & ~(size_t)255;
        return r;
    };
    float* partial = (float*)alloc((size_t)B * 32 * D * 4);
    float* Gl      = (float*)alloc((size_t)B * D * 4);
    float* Gg      = (float*)alloc(256);
    float* cr      = (float*)alloc(ND * 4);
    float* ci      = (float*)alloc(ND * 4);
    float* Xf      = (float*)alloc(ND * 4);        // gate (f32) -> wave (f32)
    __bf16* Y1     = (__bf16*)alloc(ND * 2);       // amp -> rp
    __bf16* Y2     = (__bf16*)alloc(ND * 2);       // ph  -> ip
    __bf16* Y3     = (__bf16*)alloc(ND * 2);       // gr  -> outn
    __bf16* Y4     = (__bf16*)alloc(ND * 2);       // gi
    __bf16* Hb     = (__bf16*)alloc(ND * 4 * 2);   // hr / hi / t
    // transposed bf16 weights [N,K]
    __bf16* WaT    = (__bf16*)alloc((size_t)D * D * 2);
    __bf16* WpT    = (__bf16*)alloc((size_t)D * D * 2);
    __bf16* w1rT   = (__bf16*)alloc((size_t)4 * D * D * 2);
    __bf16* w2rT   = (__bf16*)alloc((size_t)4 * D * D * 2);
    __bf16* w1iT   = (__bf16*)alloc((size_t)4 * D * D * 2);
    __bf16* w2iT   = (__bf16*)alloc((size_t)4 * D * D * 2);
    __bf16* WeT1   = (__bf16*)alloc((size_t)D * D * 2);   // top D rows of We, transposed
    __bf16* WeT2   = (__bf16*)alloc((size_t)D * D * 2);   // bottom D rows
    __bf16* wf1T   = (__bf16*)alloc((size_t)2 * D * D * 2);
    __bf16* wf2T   = (__bf16*)alloc((size_t)2 * D * D * 2);

    const int PW_BLOCKS = (int)(ND / 256);

    // 0) weights -> bf16, transposed to [N,K]
    cvtT(stream, Wa,  WaT,  D, D);
    cvtT(stream, Wp,  WpT,  D, D);
    cvtT(stream, w1r, w1rT, D, 4 * D);
    cvtT(stream, w2r, w2rT, 4 * D, D);
    cvtT(stream, w1i, w1iT, D, 4 * D);
    cvtT(stream, w2i, w2iT, 4 * D, D);
    cvtT(stream, We,                    WeT1, D, D);
    cvtT(stream, We + (size_t)D * D,    WeT2, D, D);
    cvtT(stream, wf1, wf1T, D, 2 * D);
    cvtT(stream, wf2, wf2T, 2 * D, D);

    // 1) G reductions
    k_gsq_partial<<<B * 32, 256, 0, stream>>>(x, partial, S, D);
    k_gsq_final<<<B, 256, 0, stream>>>(partial, Gl, Gg, D);

    // 2) wave representation
    k_wave<<<PW_BLOCKS, 256, 0, stream>>>(x, Gg, Gl, cr, ci, Y1, Y2, S, D);

    // 3) gate = sigmoid(amp@Wa + ph@Wp + biases) -> f32
    gemm(stream, Y1, WaT, Y2, WpT, ba, bp, gate_bias, Xf, M, D, D, D, EPI_SIGMOID, 0);

    // 4) gr/gi -> bf16
    k_gate_mul<<<PW_BLOCKS, 256, 0, stream>>>(Xf, cr, ci, Y3, Y4);

    // 5) real FFN
    gemm(stream, Y3, w1rT, nullptr, nullptr, b1r, nullptr, nullptr, Hb, M, 4 * D, D, 0, EPI_GELU, 1);
    gemm(stream, Hb, w2rT, nullptr, nullptr, b2r, nullptr, nullptr, Y1, M, D, 4 * D, 0, EPI_NONE, 1);

    // 6) imag FFN
    gemm(stream, Y4, w1iT, nullptr, nullptr, b1i, nullptr, nullptr, Hb, M, 4 * D, D, 0, EPI_GELU, 1);
    gemm(stream, Hb, w2iT, nullptr, nullptr, b2i, nullptr, nullptr, Y2, M, D, 4 * D, 0, EPI_NONE, 1);

    // 7) wave = [rp, ip] @ We + be -> f32
    gemm(stream, Y1, WeT1, Y2, WeT2, be, nullptr, nullptr, Xf, M, D, D, D, EPI_NONE, 0);

    // 8) RMSNorm -> bf16
    k_rmsnorm<<<M, 256, 0, stream>>>(Xf, g_rms, Y3, D);

    // 9) block FFN -> y (f32)
    gemm(stream, Y3, wf1T, nullptr, nullptr, bf1, nullptr, nullptr, Hb, M, 2 * D, D, 0, EPI_GELU, 1);
    gemm(stream, Hb, wf2T, nullptr, nullptr, bf2, nullptr, nullptr, y,  M, D, 2 * D, 0, EPI_NONE, 0);
}
